// ScaledDotProductAttention_86208583565785
// MI455X (gfx1250) — compile-verified, matrix operand order checked by reading, not run on hardware
//
#include <hip/hip_runtime.h>
#include <hip/hip_bf16.h>

typedef __attribute__((ext_vector_type(16))) _Float16 v16h;
typedef __attribute__((ext_vector_type(8)))  float    v8f;

#define B_  4
#define H_  8
#define L_  2048
#define D_  64
#define P_  4
#define KS  229          // 228 strided keys (step 9) + last key (2047)
#define LD  131072       // L_*D_ per (b,h) slab

// ---------------------------------------------------------------------------
// Scrambled-reshape projection, reduced to a flat conv over the (l,d) slab.
// out[l2,d2] = bias + sum_c W[c] * X[l2*64 + q + r - (f-1)],
//   q=(c*64+d2)/f, r=(c*64+d2)%f, valid iff (l2%32)!=0 || q+r>=f-1
// ---------------------------------------------------------------------------
__device__ __forceinline__ float proj_elem(const float* __restrict__ X,
                                           const float* __restrict__ w,
                                           float bias, int f, int l2, int d2) {
  const int base = l2 << 6;
  const int l2in = l2 & 31;
  float acc = bias;
  for (int c = 0; c < f; ++c) {
    int A = (c << 6) + d2;
    int q = A / f;
    int r = A - q * f;
    if (l2in || (q + r >= f - 1))
      acc += w[c] * X[base + q + r - (f - 1)];
  }
  return acc;
}

__device__ __forceinline__ const float* pick_w(int b2, const float* w0, const float* w1,
                                               const float* w2, const float* w3) {
  return b2 == 0 ? w0 : b2 == 1 ? w1 : b2 == 2 ? w2 : w3;
}
__device__ __forceinline__ int pick_f(int b2) { return b2 == 0 ? 1 : b2 == 1 ? 3 : b2 == 2 ? 6 : 9; }

// Qp[b2][h2][p2][l2][d2] = proj_{f=F[b2]}(Q[s/8, s%8]),  s = 4*h2+p2  (scrambled stack)
__global__ void proj_q_kernel(const float* __restrict__ Q,
                              const float* __restrict__ w0, const float* __restrict__ w1,
                              const float* __restrict__ w2, const float* __restrict__ w3,
                              const float* __restrict__ bq0, const float* __restrict__ bq1,
                              const float* __restrict__ bq2, const float* __restrict__ bq3,
                              float* __restrict__ qp) {
  int idx = blockIdx.x * 256 + threadIdx.x;
  int d2 = idx & 63;
  int t  = idx >> 6;
  int l2 = t & 2047; t >>= 11;
  int p2 = t & 3;    t >>= 2;
  int h2 = t & 7;
  int b2 = t >> 3;
  int s  = 4 * h2 + p2;                 // (srcb*8+srch) == s
  const float* X = Q + ((size_t)s << 17);
  const float* w = pick_w(b2, w0, w1, w2, w3);
  float bias = b2 == 0 ? bq0[0] : b2 == 1 ? bq1[0] : b2 == 2 ? bq2[0] : bq3[0];
  qp[idx] = proj_elem(X, w, bias, pick_f(b2), l2, d2);
}

// Kp[b2][h2][p2][ks][d2], ks -> l2 = ks<228 ? 9*ks : 2047
__global__ void proj_k_kernel(const float* __restrict__ K,
                              const float* __restrict__ w0, const float* __restrict__ w1,
                              const float* __restrict__ w2, const float* __restrict__ w3,
                              const float* __restrict__ bq0, const float* __restrict__ bq1,
                              const float* __restrict__ bq2, const float* __restrict__ bq3,
                              float* __restrict__ kp) {
  int idx = blockIdx.x * 256 + threadIdx.x;
  int d2 = idx & 63;
  int t  = idx >> 6;
  int ks = t % KS;
  int u  = t / KS;                      // (b2*8+h2)*4+p2
  int p2 = u & 3;
  int h2 = (u >> 2) & 7;
  int b2 = u >> 5;
  int l2 = ks < 228 ? 9 * ks : 2047;
  int s  = 4 * h2 + p2;
  const float* X = K + ((size_t)s << 17);
  const float* w = pick_w(b2, w0, w1, w2, w3);
  float bias = b2 == 0 ? bq0[0] : b2 == 1 ? bq1[0] : b2 == 2 ? bq2[0] : bq3[0];
  kp[idx] = proj_elem(X, w, bias, pick_f(b2), l2, d2);
}

// Per-(b,h): Vsum[d] over all 2048 rows, Vs[d] over the 229 special rows.
__global__ void vsum_kernel(const float* __restrict__ V,
                            float* __restrict__ vsum, float* __restrict__ vs) {
  int bh = blockIdx.x, dd = threadIdx.x;
  const float* base = V + (size_t)bh * LD;
  float s1 = 0.f;
  for (int k = 0; k < L_; ++k) s1 += base[k * 64 + dd];
  float s2 = 0.f;
  for (int k = 0; k < KS; ++k) { int col = k < 228 ? 9 * k : 2047; s2 += base[col * 64 + dd]; }
  vsum[bh * 64 + dd] = s1;
  vs[bh * 64 + dd]   = s2;
}

// Build a 16x32 f16 A/B fragment half (ISA 7.12.2 16-bit layout):
// lane: M(or N)=lane%16, hi=lane/16; element j -> K = (j<8 ? hi*8+j : 16+hi*8+(j-8)) + koff
__device__ __forceinline__ v16h load_frag16(const float* __restrict__ row, int hi, int koff) {
  const float4* p4 = (const float4*)(row + koff);
  float4 x0 = p4[hi * 2], x1 = p4[hi * 2 + 1], x2 = p4[4 + hi * 2], x3 = p4[4 + hi * 2 + 1];
  v16h a;
  a[0]=(_Float16)x0.x; a[1]=(_Float16)x0.y; a[2]=(_Float16)x0.z; a[3]=(_Float16)x0.w;
  a[4]=(_Float16)x1.x; a[5]=(_Float16)x1.y; a[6]=(_Float16)x1.z; a[7]=(_Float16)x1.w;
  a[8]=(_Float16)x2.x; a[9]=(_Float16)x2.y; a[10]=(_Float16)x2.z; a[11]=(_Float16)x2.w;
  a[12]=(_Float16)x3.x; a[13]=(_Float16)x3.y; a[14]=(_Float16)x3.z; a[15]=(_Float16)x3.w;
  return a;
}

// ---------------------------------------------------------------------------
// Fused attention: one wave32 per 16-query tile of one (b,h).
//  - scores (WMMA f16, f32 acc) over 229 keys for each of 4 "p" slices
//  - per-p softmax over 229, max over p  -> am
//  - final 2048-wide softmax is closed-form (1819 constant columns)
//  - context via WMMA over the 229 special keys + rank-1 correction
//  - stream attn_f rows with B128 stores (dominant HBM traffic: 537 MB once)
// ---------------------------------------------------------------------------
__global__ void __launch_bounds__(32) acat_attn_kernel(
    const float* __restrict__ qp, const float* __restrict__ kp,
    const float* __restrict__ V,  const float* __restrict__ vsum,
    const float* __restrict__ vs, float* __restrict__ out) {
  __shared__ float sc[16][232];
  __shared__ float am[16][232];
  __shared__ float rmax[16], rsum[16], rz[16];

  const int lane = threadIdx.x;
  const int ln16 = lane & 15;
  const int hi   = lane >> 4;
  const int bh   = blockIdx.x >> 7;     // 128 tiles of 16 rows per (b,h)
  const int tile = blockIdx.x & 127;
  const int qr0  = tile * 16;
  const float E0 = __expf(1.0f / 2048.0f);

  float* out_ctx = out;
  float* out_att = out + (size_t)B_ * H_ * L_ * D_;

  for (int p2 = 0; p2 < P_; ++p2) {
    // A fragments: Q tile rows (query qr0+ln16), K-dim 0..31 and 32..63
    const float* qrow = qp + ((size_t)((bh * 4 + p2) * L_ + qr0 + ln16) << 6);
    v16h a0 = load_frag16(qrow, hi, 0);
    v16h a1 = load_frag16(qrow, hi, 32);

    for (int nt = 0; nt < 15; ++nt) {          // 15*16 = 240 >= 229 keys
      int key = nt * 16 + ln16;
      bool kvalid = key < KS;
      const float* krow = kp + ((size_t)((bh * 4 + p2) * KS + (kvalid ? key : 0)) << 6);
      v16h zf = {};
      v16h b0 = kvalid ? load_frag16(krow, hi, 0)  : zf;
      v16h b1 = kvalid ? load_frag16(krow, hi, 32) : zf;
      v8f c = {};
      c = __builtin_amdgcn_wmma_f32_16x16x32_f16(false, a0, false, b0, (short)0, c, false, false);
      c = __builtin_amdgcn_wmma_f32_16x16x32_f16(false, a1, false, b1, (short)0, c, false, false);
      if (kvalid) {
        int col = key < 228 ? 9 * key : 2047;
#pragma unroll
        for (int v = 0; v < 8; ++v) {          // C layout: N=lane%16, M=v+8*hi
          int m = v + 8 * hi;
          float val = c[v] * 0.125f;           // / sqrt(64)
          sc[m][key] = (col > qr0 + m) ? -1.0e9f : val;   // causal mask
        }
      }
    }
    __syncthreads();
    if (lane < 16) {                            // per-row softmax stats over 229
      float mx = -3.0e38f;
      for (int k = 0; k < KS; ++k) mx = fmaxf(mx, sc[lane][k]);
      float s = 0.f;
      for (int k = 0; k < KS; ++k) s += __expf(sc[lane][k] - mx);
      rmax[lane] = mx;
      rsum[lane] = 1.0f / s;
    }
    __syncthreads();
    for (int r = 0; r < 16; ++r)
      for (int k = lane; k < KS; k += 32) {
        float smv = __expf(sc[r][k] - rmax[r]) * rsum[r];
        am[r][k] = (p2 == 0) ? smv : fmaxf(am[r][k], smv);
      }
    __syncthreads();
  }

  // ea = exp(attn); Z = 1819*e^(1/2048) + sum(ea)   (closed-form outer softmax)
  if (lane < 16) {
    float z = 0.f;
    for (int k = 0; k < KS; ++k) { float e = __expf(am[lane][k]); sc[lane][k] = e; z += e; }
    rz[lane] = 1819.0f * E0 + z;
  }
  __syncthreads();

  // context = ( E0*(Vsum - Vs) + ea(16xKS) @ Vspecial(KSx64) ) / Z   via WMMA
  v16h af[8];
#pragma unroll
  for (int ch = 0; ch < 8; ++ch) {             // 8 K-chunks of 32 cover KS (pad 0)
    v16h a;
#pragma unroll
    for (int j = 0; j < 16; ++j) {
      int kk = ch * 32 + ((j < 8) ? (hi * 8 + j) : (16 + hi * 8 + (j - 8)));
      a[j] = (kk < KS) ? (_Float16)sc[ln16][kk] : (_Float16)0.0f;
    }
    af[ch] = a;
  }
  const float* vbase = V + (size_t)bh * LD;
  for (int nt = 0; nt < 4; ++nt) {             // 4 tiles of 16 over d=64
    int dd = nt * 16 + ln16;
    v8f c = {};
    for (int ch = 0; ch < 8; ++ch) {
      v16h b;
#pragma unroll
      for (int j = 0; j < 16; ++j) {
        int kk = ch * 32 + ((j < 8) ? (hi * 8 + j) : (16 + hi * 8 + (j - 8)));
        float vv = 0.f;
        if (kk < KS) { int col = kk < 228 ? 9 * kk : 2047; vv = vbase[col * 64 + dd]; }
        b[j] = (_Float16)vv;
      }
      c = __builtin_amdgcn_wmma_f32_16x16x32_f16(false, af[ch], false, b, (short)0, c, false, false);
    }
    float vdiff = E0 * (vsum[bh * 64 + dd] - vs[bh * 64 + dd]);
#pragma unroll
    for (int v = 0; v < 8; ++v) {
      int m = v + 8 * hi;
      out_ctx[((size_t)(bh * L_ + qr0 + m) << 6) + dd] = (vdiff + c[v]) / rz[m];
    }
  }

  // attn_f rows: constant E0/Z except special columns; B128 (float4) stores.
  for (int r = 0; r < 16; ++r) {
    float invZ = 1.0f / rz[r];
    float regv = E0 * invZ;
    float* rowp = out_att + ((size_t)(bh * L_ + qr0 + r) << 11);
    for (int c0 = lane * 4; c0 < L_; c0 += 128) {
      float4 o;
#pragma unroll
      for (int i = 0; i < 4; ++i) {
        int col = c0 + i;
        int q9 = col / 9;
        float val;
        if (col - q9 * 9 == 0)  val = sc[r][q9] * invZ;
        else if (col == 2047)   val = sc[r][228] * invZ;
        else                    val = regv;
        ((float*)&o)[i] = val;
      }
      *(float4*)(rowp + c0) = o;
    }
  }
}

extern "C" void kernel_launch(void* const* d_in, const int* in_sizes, int n_in,
                              void* d_out, int out_size, void* d_ws, size_t ws_size,
                              hipStream_t stream) {
  (void)in_sizes; (void)n_in; (void)out_size; (void)ws_size;
  const float* Q  = (const float*)d_in[0];
  const float* K  = (const float*)d_in[1];
  const float* V  = (const float*)d_in[2];
  const float* w0 = (const float*)d_in[3];
  const float* w1 = (const float*)d_in[4];
  const float* w2 = (const float*)d_in[5];
  const float* w3 = (const float*)d_in[6];
  const float* q0 = (const float*)d_in[7];
  const float* q1 = (const float*)d_in[8];
  const float* q2 = (const float*)d_in[9];
  const float* q3 = (const float*)d_in[10];

  float* ws  = (float*)d_ws;
  float* qp  = ws;                                         // 16,777,216 f32
  float* kp  = qp + (size_t)B_ * H_ * P_ * L_ * D_;        //  1,875,968 f32
  float* vsm = kp + (size_t)B_ * H_ * P_ * KS * D_;        //      2,048 f32
  float* vsp = vsm + B_ * H_ * D_;                         //      2,048 f32

  proj_q_kernel<<<(B_ * H_ * P_ * L_ * D_) / 256, 256, 0, stream>>>(
      Q, w0, w1, w2, w3, q0, q1, q2, q3, qp);
  proj_k_kernel<<<(B_ * H_ * P_ * KS * D_) / 256, 256, 0, stream>>>(
      K, w0, w1, w2, w3, q0, q1, q2, q3, kp);
  vsum_kernel<<<B_ * H_, D_, 0, stream>>>(V, vsm, vsp);
  acat_attn_kernel<<<B_ * H_ * (L_ / 16), 32, 0, stream>>>(
      qp, kp, V, vsm, vsp, (float*)d_out);
}
